// CollageDecoder_4174708212009
// MI455X (gfx1250) — matrix operand order repaired and checked by reading, not compiled, not run on hardware
//
#include <hip/hip_runtime.h>
#include <hip/hip_bf16.h>

typedef __attribute__((ext_vector_type(16))) _Float16 v16h;
typedef __attribute__((ext_vector_type(8)))  float    v8f;

#define KDIM 256
#define NT   147456      // 3*1024*16*3
#define NR   1024
#define ND   16
#define NC   3
#define BS   8
#define RES  256

// ---------------------------------------------------------------------------
// Kernel 1: pe_mean[b,k] = mean over 64 spatial positions of patch_emb
// grid: 8 blocks (b), 256 threads (k)
__global__ __launch_bounds__(256) void pemean_k(const float* __restrict__ pe,
                                                float* __restrict__ pm) {
  const int b = blockIdx.x, k = threadIdx.x;
  float s = 0.f;
  for (int j = 0; j < 64; ++j) s += pe[((size_t)b * 64 + j) * 256 + k];
  pm[b * 256 + k] = s * (1.0f / 64.0f);
}

// ---------------------------------------------------------------------------
// Kernel 2: theta[b,t] = pe_mean[b,:] . w[:,t] + bias[t]  via WMMA f16->f32
// One wave computes 8 N-tiles of 16 columns (M padded 8->16, K = 8 x 32).
// grid: 144 blocks x 256 threads = 1152 waves; 1152*128 = 147456 columns.
__global__ __launch_bounds__(256) void gemm_theta(const float* __restrict__ pm,
                                                  const float* __restrict__ w,
                                                  const float* __restrict__ bias,
                                                  float* __restrict__ theta) {
  const int lane  = threadIdx.x & 31;
  const int wave  = blockIdx.x * (blockDim.x >> 5) + (threadIdx.x >> 5);
  const int nbase = wave * 128;

  // --- Preload A (pe_mean, padded to 16 rows) in this code's contiguous-k A 16x32 layout (B uses the same k order):
  // lanes 0-15: M=lane, halfs h -> K = h
  // lanes16-31: M=lane-16, halfs h -> K = 16+h
  const int m = lane & 15;
  v16h A[8];
  for (int kt = 0; kt < 8; ++kt) {
    v16h a{};
    for (int h = 0; h < 16; ++h) {
      int kk = (lane < 16) ? h : 16 + h;
      float v = (m < BS) ? pm[m * KDIM + kt * 32 + kk] : 0.0f;
      a[h] = (_Float16)v;
    }
    A[kt] = a;
  }

  // --- B layout (32x16, K x N): lanes 0-15 hold K=0..15 at N=lane,
  // lanes 16-31 hold K=16..31 at N=lane-16; half index = K % 16.
  const int ncol  = lane & 15;
  const int khalf = lane >> 4;

  for (int t = 0; t < 8; ++t) {
    const int n0 = nbase + t * 16;
    v8f C = {};
    for (int kt = 0; kt < 8; ++kt) {
      const int kb = kt * 32 + khalf * 16;
      const float* wp = w + (size_t)kb * NT + n0 + ncol;
      __builtin_prefetch(wp + 16, 0, 0);   // next tile's column, speculative
      v16h B{};
      for (int h = 0; h < 16; ++h) B[h] = (_Float16)wp[(size_t)h * NT];
      C = __builtin_amdgcn_wmma_f32_16x16x32_f16(false, A[kt], false, B,
                                                 (short)0, C, false, false);
    }
    // D layout: VGPR j, lane L: M = j + 8*(L>=16), N = L&15. Rows 8..15 are pad.
    if (lane < 16) {
      const int n = n0 + lane;
      const float bv = bias[n];
      for (int j = 0; j < 8; ++j)
        theta[(size_t)j * NT + n] = C[j] + bv;
    }
  }
}

// ---------------------------------------------------------------------------
// Kernel 3: softmax over r (1024) for fixed (b,d,c); mulw/addw = tanh(.)*softmax
// grid: 384 blocks = (b,d,c); 256 threads, 4 r's per thread.
__global__ __launch_bounds__(256) void postproc_k(const float* __restrict__ theta,
                                                  float* __restrict__ mulw,
                                                  float* __restrict__ addw) {
  const int b = blockIdx.x / (ND * NC);
  const int rem = blockIdx.x % (ND * NC);
  const int d = rem / NC, c = rem % NC;
  __shared__ float red[256];
  const int t = threadIdx.x;

  float mixv[4];
  float mx = -3.0e38f;
  for (int i = 0; i < 4; ++i) {
    const int r = t + i * 256;
    const size_t idx = ((size_t)(b * 3072 + 2048 + r) * ND + d) * NC + c;
    mixv[i] = theta[idx];
    mx = fmaxf(mx, mixv[i]);
  }
  red[t] = mx; __syncthreads();
  for (int s = 128; s > 0; s >>= 1) {
    if (t < s) red[t] = fmaxf(red[t], red[t + s]);
    __syncthreads();
  }
  mx = red[0]; __syncthreads();

  float sum = 0.f;
  for (int i = 0; i < 4; ++i) { mixv[i] = __expf(mixv[i] - mx); sum += mixv[i]; }
  red[t] = sum; __syncthreads();
  for (int s = 128; s > 0; s >>= 1) {
    if (t < s) red[t] += red[t + s];
    __syncthreads();
  }
  const float inv = 1.0f / red[0];

  for (int i = 0; i < 4; ++i) {
    const int r = t + i * 256;
    const float sm = mixv[i] * inv;
    const size_t im = ((size_t)(b * 3072 + r) * ND + d) * NC + c;          // mul
    const size_t ia = ((size_t)(b * 3072 + 1024 + r) * ND + d) * NC + c;   // add
    const size_t o  = ((size_t)(b * NR + r) * (ND * NC)) + d * NC + c;
    mulw[o] = tanhf(theta[im]) * sm;
    addw[o] = tanhf(theta[ia]) * sm;
  }
}

// ---------------------------------------------------------------------------
// Kernel 4: add_sum[b,r,c] = sum_d addw ; also seeds m1 = add_sum
__global__ __launch_bounds__(256) void addsum_k(const float* __restrict__ addw,
                                                float* __restrict__ asum,
                                                float* __restrict__ m1) {
  const int i = blockIdx.x * blockDim.x + threadIdx.x;   // (b*1024+r)*3+c
  if (i >= BS * NR * NC) return;
  const int c = i % NC, br = i / NC;
  float s = 0.f;
  for (int d = 0; d < ND; ++d) s += addw[(size_t)br * (ND * NC) + d * NC + c];
  asum[i] = s;
  m1[i]   = s;
}

// ---------------------------------------------------------------------------
// Kernel 5a: pbar[b,d,c] = mean of m over the 64 range patches inside domain d
__global__ __launch_bounds__(384) void pbar_k(const float* __restrict__ m,
                                              float* __restrict__ pbar) {
  const int i = threadIdx.x;                  // b*48 + d*3 + c, 384 total
  if (i >= BS * ND * NC) return;
  const int c = i % NC, rem = i / NC, d = rem % ND, b = rem / ND;
  const int dy = d >> 2, dx = d & 3;
  float s = 0.f;
  for (int ii = 0; ii < 8; ++ii)
    for (int jj = 0; jj < 8; ++jj) {
      const int r = (dy * 8 + ii) * 32 + dx * 8 + jj;
      s += m[(size_t)(b * NR + r) * NC + c];
    }
  pbar[i] = s * (1.0f / 64.0f);
}

// Kernel 5b: m_out[b,r,c] = sum_d mulw[b,r,d,c]*pbar[b,d,c] + add_sum[b,r,c]
__global__ __launch_bounds__(256) void mstep_k(const float* __restrict__ mulw,
                                               const float* __restrict__ pbar,
                                               const float* __restrict__ asum,
                                               float* __restrict__ mout) {
  const int i = blockIdx.x * blockDim.x + threadIdx.x;
  if (i >= BS * NR * NC) return;
  const int c = i % NC, br = i / NC, b = br >> 10;
  float s = asum[i];
  for (int d = 0; d < ND; ++d)
    s += mulw[(size_t)br * (ND * NC) + d * NC + c] * pbar[(b * ND + d) * NC + c];
  mout[i] = s;
}

// ---------------------------------------------------------------------------
// Kernel 6: final render (step 10, spatial form) from m9
// out[b,y,x,c] = sum_d mulw[b,r,d,c]*m9[b, r(d, y%8, x%8), c] + add_sum[b,r,c]
__global__ __launch_bounds__(256) void render_k(const float* __restrict__ mulw,
                                                const float* __restrict__ asum,
                                                const float* __restrict__ m9,
                                                float* __restrict__ out) {
  const int i = blockIdx.x * blockDim.x + threadIdx.x;
  if (i >= BS * RES * RES * NC) return;
  const int c  = i % NC;
  const int x  = (i / NC) % RES;
  const int y  = (i / (NC * RES)) % RES;
  const int b  = i / (NC * RES * RES);
  const int h = y & 7, w = x & 7;
  const int r = (y >> 3) * 32 + (x >> 3);
  float s = asum[(size_t)(b * NR + r) * NC + c];
  for (int d = 0; d < ND; ++d) {
    const int dy = d >> 2, dx = d & 3;
    const int rs = (dy * 8 + h) * 32 + dx * 8 + w;
    s += mulw[(size_t)(b * NR + r) * (ND * NC) + d * NC + c]
       * m9[(size_t)(b * NR + rs) * NC + c];
  }
  out[i] = s;
}

// ---------------------------------------------------------------------------
extern "C" void kernel_launch(void* const* d_in, const int* in_sizes, int n_in,
                              void* d_out, int out_size, void* d_ws, size_t ws_size,
                              hipStream_t stream) {
  const float* pe   = (const float*)d_in[0];   // (8,8,8,256)
  const float* w    = (const float*)d_in[1];   // (256,147456)
  const float* bias = (const float*)d_in[2];   // (147456,)
  float* out = (float*)d_out;

  float* ws = (float*)d_ws;
  float* pm    = ws;                    // 2048
  float* theta = pm    + 2048;          // 1179648
  float* mulw  = theta + 1179648;       // 393216
  float* addw  = mulw  + 393216;        // 393216
  float* asum  = addw  + 393216;        // 24576
  float* mA    = asum  + 24576;         // 24576
  float* mB    = mA    + 24576;         // 24576
  float* pbar  = mB    + 24576;         // 384   -> ~8.2 MB total

  pemean_k<<<BS, 256, 0, stream>>>(pe, pm);
  gemm_theta<<<144, 256, 0, stream>>>(pm, w, bias, theta);
  postproc_k<<<BS * ND * NC, 256, 0, stream>>>(theta, mulw, addw);
  addsum_k<<<(BS * NR * NC + 255) / 256, 256, 0, stream>>>(addw, asum, mA);

  // m1 = add_sum; 8 recursion steps give m9 (back in mA after even # of swaps)
  float* cur = mA;
  float* nxt = mB;
  for (int s = 0; s < 8; ++s) {
    pbar_k<<<1, 384, 0, stream>>>(cur, pbar);
    mstep_k<<<(BS * NR * NC + 255) / 256, 256, 0, stream>>>(mulw, pbar, asum, nxt);
    float* tmp = cur; cur = nxt; nxt = tmp;
  }

  render_k<<<(BS * RES * RES * NC + 255) / 256, 256, 0, stream>>>(mulw, asum, cur, out);
}
